// SelfAttention_28939489640663
// MI455X (gfx1250) — matrix operand order repaired and checked by reading, not compiled, not run on hardware
//
#include <hip/hip_runtime.h>
#include <hip/hip_bf16.h>

// ---------------------------------------------------------------------------
// Self-attention (B=8, H=W=64, C=256, D=32) for gfx1250 (CDNA5, wave32, WMMA).
//   1) proj_f16_kernel<E,TR> : Q,K (row-major f16) and V (TRANSPOSED (B,C,N)
//      f16) via v_wmma_f32_16x16x32_f16.  Transposed V makes every later
//      LDS/global access 16-byte wide.
//   2) attn_kernel: flash attention, 16 Q rows x 64-column K/V chunks,
//      online softmax; row-sums of P computed by a ones-matrix WMMA instead
//      of ds_bpermute butterflies; epilogue out = gamma*attn + x.
// All offsets are 32-bit (buffers < 2^31 elements) to avoid v_mul_u64.
// ---------------------------------------------------------------------------

#define BB 8
#define NN 4096
#define CC 256
#define DD 32
#define CH 64      // columns (m) per streamed chunk

typedef __attribute__((ext_vector_type(16))) _Float16 v16h;
typedef __attribute__((ext_vector_type(8)))  _Float16 h8;
typedef __attribute__((ext_vector_type(4)))  _Float16 h4;
typedef __attribute__((ext_vector_type(8)))  float    v8f;
typedef __attribute__((ext_vector_type(4)))  float    v4f;

union Frag16 { v16h v; h8 h[2]; };

static __device__ __forceinline__ v8f wmma_f16(v16h a, v16h b, v8f c) {
  return __builtin_amdgcn_wmma_f32_16x16x32_f16(false, a, false, b,
                                                (short)0, c, false, false);
}

// ---------------------------------------------------------------------------
// Projection: Out = f16( X(R,CC) @ W(CC,E) + bias ), R = B*N.
// TR=false: Out row-major (R,E).  TR=true: Out transposed per batch (B,E,N).
// One wave computes one 16x16 output tile.
// ---------------------------------------------------------------------------
template <int E, bool TR>
__global__ __launch_bounds__(32)
void proj_f16_kernel(const float* __restrict__ X,
                     const float* __restrict__ W,
                     const float* __restrict__ bias,
                     _Float16* __restrict__ Out)
{
  const int r0   = blockIdx.x * 16;
  const int c0   = blockIdx.y * 16;
  const int lane = threadIdx.x;          // 0..31
  const int g    = lane >> 4;            // half-wave group
  const int ln   = lane & 15;

  __shared__ __align__(16) _Float16 Xlds[16 * 40];   // [row][k]
  __shared__ __align__(16) _Float16 Wlds[16 * 40];   // [col][k]

  v8f acc = {};

  for (int kk = 0; kk < CC; kk += 32) {
    // Stage X tile (16 x 32) : float4 loads, packed b64 LDS stores.
#pragma unroll
    for (int i = 0; i < 4; ++i) {
      int seg = i * 32 + lane;           // 0..127
      int row = seg >> 3;
      int k4  = (seg & 7) * 4;
      v4f xv = *(const v4f*)(X + (r0 + row) * CC + kk + k4);
      h4 xh;
#pragma unroll
      for (int j = 0; j < 4; ++j) xh[j] = (_Float16)xv[j];
      *(h4*)&Xlds[row * 40 + k4] = xh;
    }
    // Stage W tile (32 k x 16 c) k-contiguous per column (transpose scatter).
#pragma unroll
    for (int i = 0; i < 4; ++i) {
      int seg = i * 32 + lane;           // 0..127
      int k   = seg >> 2;
      int c4  = (seg & 3) * 4;
      v4f wv = *(const v4f*)(W + (kk + k) * E + c0 + c4);
#pragma unroll
      for (int j = 0; j < 4; ++j)
        Wlds[(c4 + j) * 40 + k] = (_Float16)wv[j];
    }
    __syncthreads();

    Frag16 a, b;
    a.h[0] = *(const h8*)&Xlds[ln * 40 + 8 * g];
    a.h[1] = *(const h8*)&Xlds[ln * 40 + 16 + 8 * g];
    b.h[0] = *(const h8*)&Wlds[ln * 40 + 8 * g];
    b.h[1] = *(const h8*)&Wlds[ln * 40 + 16 + 8 * g];

    acc = wmma_f16(a.v, b.v, acc);
    __syncthreads();
  }

  const float bc = bias[c0 + ln];
  if (TR) {
    // Transposed store: lane's 8 rows are contiguous in N -> one b128 store.
    int rowg = r0 + 8 * g;               // first global row for this lane
    int bidx = rowg >> 12;               // / NN
    int nb   = rowg & (NN - 1);
    h8 o;
#pragma unroll
    for (int r = 0; r < 8; ++r) o[r] = (_Float16)(acc[r] + bc);
    *(h8*)(Out + bidx * (E * NN) + (c0 + ln) * NN + nb) = o;
  } else {
#pragma unroll
    for (int r = 0; r < 8; ++r) {
      int row = r + 8 * g;               // C-frag: VGPR r <-> M = r + 8g
      Out[(r0 + row) * E + c0 + ln] = (_Float16)(acc[r] + bc);
    }
  }
}

// ---------------------------------------------------------------------------
// Flash attention, one wave per (16-row tile, batch).
//   Kview[d][m] = Kflat[d*N+m]  (faithful no-transpose reshape)
//   S(16x64) = Q(16x32) x Kview chunk ; online softmax ;
//   rowsum(P) via ones-matrix WMMA ;
//   O(16x256) += P(16x64) x V(64x256)  using transposed V (C,N).
// ---------------------------------------------------------------------------
__global__ __launch_bounds__(32)
void attn_kernel(const _Float16* __restrict__ Qh,   // (B, N, D)
                 const _Float16* __restrict__ Kh,   // (B, N*D) flat
                 const _Float16* __restrict__ Vt,   // (B, C, N) transposed
                 const float*    __restrict__ x,    // (B, N, C)
                 const float*    __restrict__ gamma,
                 float*          __restrict__ out)  // (B, N, C)
{
  const int n0   = blockIdx.x * 16;
  const int b    = blockIdx.y;
  const int lane = threadIdx.x;
  const int g    = lane >> 4;
  const int ln   = lane & 15;

  const _Float16* Qb = Qh + b * (NN * DD);
  const _Float16* Kb = Kh + b * (NN * DD);
  const _Float16* Vb = Vt + b * (CC * NN);

  __shared__ __align__(16) _Float16 Klds[CH * 40];   // [m][d]
  __shared__ __align__(16) _Float16 Vlds[CC * 72];   // [c][m], m=0..63
  __shared__ __align__(16) _Float16 Plds[16 * 72];   // [row][m]

  // Q tile A-fragment (two contiguous 16B runs per lane).
  Frag16 qa;
  qa.h[0] = *(const h8*)(Qb + (n0 + ln) * DD + 8 * g);
  qa.h[1] = *(const h8*)(Qb + (n0 + ln) * DD + 16 + 8 * g);

  // All-ones B fragment: rowsum(P) = P x Ones via the matrix unit.
  Frag16 onesf;
#pragma unroll
  for (int j = 0; j < 8; ++j) { onesf.h[0][j] = (_Float16)1.0f;
                                onesf.h[1][j] = (_Float16)1.0f; }

  v8f O[16];
#pragma unroll
  for (int t = 0; t < 16; ++t) O[t] = (v8f){};
  float m_old[8], l[8];
#pragma unroll
  for (int r = 0; r < 8; ++r) { m_old[r] = -1e30f; l[r] = 0.0f; }

  for (int m0 = 0; m0 < NN; m0 += CH) {
    if (m0 + CH < NN) {  // global_prefetch_b8 next chunk
      __builtin_prefetch((const void*)(Kb + lane * NN + m0 + CH), 0, 1);
      __builtin_prefetch((const void*)(Vb + (lane * 8) * NN + m0 + CH), 0, 1);
    }

    // ---- Stage K chunk: lane owns row d=lane; wide loads, b16 scatter.
    h8 kr[8];
#pragma unroll
    for (int u = 0; u < 8; ++u)
      kr[u] = *(const h8*)(Kb + lane * NN + m0 + u * 8);
#pragma unroll
    for (int u = 0; u < 8; ++u) {
#pragma unroll
      for (int j = 0; j < 8; ++j)
        Klds[(u * 8 + j) * 40 + lane] = kr[u][j];
    }

    // ---- Stage V chunk: transposed layout makes this a pure b128 copy.
#pragma unroll 8
    for (int i = 0; i < 64; ++i) {
      int s  = i * 32 + lane;            // 0..2047
      int c  = s >> 3;
      int m8 = (s & 7) * 8;
      *(h8*)&Vlds[c * 72 + m8] = *(const h8*)(Vb + c * NN + m0 + m8);
    }
    __syncthreads();

    // ---- 4 score tiles S(16x16), cols m0 + t*16 ...
    v8f S[4];
#pragma unroll
    for (int t = 0; t < 4; ++t) {
      Frag16 kb;
      kb.h[0] = *(const h8*)&Klds[(t * 16 + ln) * 40 + 8 * g];
      kb.h[1] = *(const h8*)&Klds[(t * 16 + ln) * 40 + 16 + 8 * g];
      v8f z = {};
      S[t] = wmma_f16(qa.v, kb.v, z);
    }

    // ---- Row max across 16 lanes of each half-wave (rows M = r + 8g).
    float mx[8];
#pragma unroll
    for (int r = 0; r < 8; ++r)
      mx[r] = fmaxf(fmaxf(S[0][r], S[1][r]), fmaxf(S[2][r], S[3][r]));
#pragma unroll
    for (int off = 1; off < 16; off <<= 1) {
#pragma unroll
      for (int r = 0; r < 8; ++r)
        mx[r] = fmaxf(mx[r], __shfl_xor(mx[r], off, 32));
    }

    float alpha[8];
#pragma unroll
    for (int r = 0; r < 8; ++r) {
      float m_new = fmaxf(m_old[r], mx[r]);
      alpha[r]    = __expf(m_old[r] - m_new);
      m_old[r]    = m_new;
#pragma unroll
      for (int t = 0; t < 4; ++t) S[t][r] = __expf(S[t][r] - m_new);
    }

    // ---- Rescale running output (once per 64 columns).
#pragma unroll
    for (int t = 0; t < 16; ++t) {
#pragma unroll
      for (int r = 0; r < 8; ++r) O[t][r] *= alpha[r];
    }

    // ---- Bounce P (C-layout) through LDS into A-layout.
#pragma unroll
    for (int t = 0; t < 4; ++t) {
#pragma unroll
      for (int r = 0; r < 8; ++r)
        Plds[(r + 8 * g) * 72 + t * 16 + ln] = (_Float16)S[t][r];
    }
    __syncthreads();

    Frag16 pa[2];
#pragma unroll
    for (int mh = 0; mh < 2; ++mh) {
      pa[mh].h[0] = *(const h8*)&Plds[ln * 72 + mh * 32 + 8 * g];
      pa[mh].h[1] = *(const h8*)&Plds[ln * 72 + mh * 32 + 16 + 8 * g];
    }

    // ---- Row sums of P via ones-WMMA: every column of D holds rowsum(M).
    {
      v8f z = {};
      v8f rs = wmma_f16(pa[0].v, onesf.v, z);
      rs     = wmma_f16(pa[1].v, onesf.v, rs);
#pragma unroll
      for (int r = 0; r < 8; ++r) l[r] = l[r] * alpha[r] + rs[r];
    }

    // ---- O(16x256) += P(16x64) x V(64x256): 16 tiles x 2 k-halves.
#pragma unroll
    for (int t = 0; t < 16; ++t) {
#pragma unroll
      for (int mh = 0; mh < 2; ++mh) {
        Frag16 vb;
        vb.h[0] = *(const h8*)&Vlds[(t * 16 + ln) * 72 + mh * 32 + 8 * g];
        vb.h[1] = *(const h8*)&Vlds[(t * 16 + ln) * 72 + mh * 32 + 16 + 8 * g];
        O[t] = wmma_f16(pa[mh].v, vb.v, O[t]);
      }
    }
    __syncthreads();
  }

  // ---- Epilogue: out = gamma * (O / l) + x.
  const float gm = gamma[0];
#pragma unroll
  for (int r = 0; r < 8; ++r) {
    const float inv = 1.0f / l[r];
    const int base = (b * NN + n0 + r + 8 * g) * CC + ln;
    const float* xr   = x   + base;
    float*       orow = out + base;
#pragma unroll
    for (int t = 0; t < 16; ++t)
      orow[t * 16] = gm * (O[t][r] * inv) + xr[t * 16];
  }
}

// ---------------------------------------------------------------------------

extern "C" void kernel_launch(void* const* d_in, const int* in_sizes, int n_in,
                              void* d_out, int out_size, void* d_ws, size_t ws_size,
                              hipStream_t stream) {
  const float* x     = (const float*)d_in[0];
  const float* Wq    = (const float*)d_in[1];
  const float* bq    = (const float*)d_in[2];
  const float* Wk    = (const float*)d_in[3];
  const float* bk    = (const float*)d_in[4];
  const float* Wv    = (const float*)d_in[5];
  const float* bv    = (const float*)d_in[6];
  const float* gamma = (const float*)d_in[7];
  float* out = (float*)d_out;

  char* ws = (char*)d_ws;
  _Float16* Qh = (_Float16*)(ws);                               // 2 MB
  _Float16* Kh = (_Float16*)(ws + (size_t)2 * 1024 * 1024);     // 2 MB
  _Float16* Vt = (_Float16*)(ws + (size_t)4 * 1024 * 1024);     // 16 MB (B,C,N)

  dim3 blk(32);

  proj_f16_kernel<DD, false><<<dim3((BB * NN) / 16, DD / 16), blk, 0, stream>>>(x, Wq, bq, Qh);
  proj_f16_kernel<DD, false><<<dim3((BB * NN) / 16, DD / 16), blk, 0, stream>>>(x, Wk, bk, Kh);
  proj_f16_kernel<CC, true ><<<dim3((BB * NN) / 16, CC / 16), blk, 0, stream>>>(x, Wv, bv, Vt);

  attn_kernel<<<dim3(NN / 16, BB), blk, 0, stream>>>(Qh, Kh, Vt, x, gamma, out);
}
